// MultiHeadSelfAttention_89472758710800
// MI455X (gfx1250) — matrix operand order repaired and checked by reading, not compiled, not run on hardware
//
#include <hip/hip_runtime.h>

// MI455X / gfx1250: wave32, WMMA 16x16x32 f16 -> f32
typedef __attribute__((ext_vector_type(16))) _Float16 v16h;
typedef __attribute__((ext_vector_type(8)))  float    v8f;

#define S_DIM 2048
#define E_DIM 2048
#define H_DIM 16
#define D_DIM 128

union Frag16 { uint4 u[2]; v16h v; };

__device__ __forceinline__ v16h load_frag_2x16B(const _Float16* p0, const _Float16* p1) {
  Frag16 f;
  f.u[0] = *reinterpret_cast<const uint4*>(p0);
  f.u[1] = *reinterpret_cast<const uint4*>(p1);
  return f.v;
}
__device__ __forceinline__ v16h load_frag_32B(const _Float16* p) {
  Frag16 f;
  const uint4* q = reinterpret_cast<const uint4*>(p);
  f.u[0] = q[0];
  f.u[1] = q[1];
  return f.v;
}
__device__ __forceinline__ v8f wmma_f16(v16h a, v16h b, v8f c) {
  return __builtin_amdgcn_wmma_f32_16x16x32_f16(false, a, false, b, (short)0, c, false, false);
}

// ---------------------------------------------------------------------------
// Generic GEMM: C[M,N] = A[M,K] * B[K,N] + bias[N]
// A: TA (float or _Float16) row-major; B: float row-major (converted to f16);
// C: f32 (OUT_F32) or f16.
// Block = 128x128, 256 threads = 8 waves, wave tile = 32x64 (2x4 WMMA tiles).
// K staged 64 at a time into ping-pong LDS buffers with register prefetch:
// one barrier and 16 WMMAs per staged tile; global loads for tile t+1 are
// issued before the WMMA block so they overlap with matrix math; tile t+2 is
// prefetched into cache with global_prefetch_b8.
// ---------------------------------------------------------------------------
template <typename TA, bool OUT_F32>
__global__ __launch_bounds__(256, 1)
void gemm_wmma(const TA* __restrict__ A, const float* __restrict__ B,
               const float* __restrict__ bias, void* __restrict__ Cv,
               int M, int N, int K) {
  __shared__ __align__(16) _Float16 As[2][128][72];  // [buf][m][k]  +8 pad
  __shared__ __align__(16) _Float16 Bt[2][128][72];  // [buf][n][k]  transposed

  const int tid   = threadIdx.x;
  const int lane  = tid & 31;
  const int wave  = tid >> 5;
  const int waveM = (wave >> 1) * 32;  // 0,32,64,96
  const int waveN = (wave & 1) * 64;   // 0,64
  const int bM    = blockIdx.y * 128;
  const int bN    = blockIdx.x * 128;

  const int mlane = lane & 15;
  const int kbA   = (lane >> 4) * 8;   // A-frag K base within half
  const int kbB   = (lane >> 4) * 16;  // unused: B frags use kbA

  const v8f vzero = {0.f, 0.f, 0.f, 0.f, 0.f, 0.f, 0.f, 0.f};
  v8f acc[2][4];
#pragma unroll
  for (int mi = 0; mi < 2; ++mi)
#pragma unroll
    for (int ni = 0; ni < 4; ++ni) acc[mi][ni] = vzero;

  const int aRow = tid >> 1;           // 0..127
  const int aCol = (tid & 1) * 32;     // 0,32
  const int bRow = tid >> 2;           // 0..63
  const int bCol = (tid & 3) * 32;     // 0..96

  TA    aReg[32];
  float bReg[32];

  auto loadTiles = [&](int kt) {
    const TA* asrc = A + (size_t)(bM + aRow) * K + kt + aCol;
#pragma unroll
    for (int i = 0; i < 32; ++i) aReg[i] = asrc[i];
    const float* bsrc = B + (size_t)(kt + bRow) * N + bN + bCol;
#pragma unroll
    for (int i = 0; i < 32; ++i) bReg[i] = bsrc[i];
  };
  auto storeTiles = [&](int buf) {
#pragma unroll
    for (int i = 0; i < 32; ++i) As[buf][aRow][aCol + i] = (_Float16)aReg[i];
#pragma unroll
    for (int i = 0; i < 32; ++i) Bt[buf][bCol + i][bRow] = (_Float16)bReg[i];
  };

  loadTiles(0);
  storeTiles(0);
  __syncthreads();

  const int NT = K >> 6;  // K / 64
  for (int t = 0; t < NT; ++t) {
    const int cur = t & 1;
    if (t + 1 < NT) loadTiles((t + 1) * 64);  // overlap with WMMA below
    if (t + 2 < NT) {
      __builtin_prefetch(A + (size_t)(bM + aRow) * K + (t + 2) * 64 + aCol, 0, 3);
      __builtin_prefetch(B + (size_t)((t + 2) * 64 + bRow) * N + bN + bCol, 0, 3);
    }

#pragma unroll
    for (int f = 0; f < 2; ++f) {  // two k-steps of 32 inside the staged 64
      v16h afrag[2], bfrag[4];
#pragma unroll
      for (int mi = 0; mi < 2; ++mi) {
        const _Float16* p = &As[cur][waveM + mi * 16 + mlane][f * 32 + kbA];
        afrag[mi] = load_frag_2x16B(p, p + 16);
      }
#pragma unroll
      for (int ni = 0; ni < 4; ++ni)
        bfrag[ni] = load_frag_2x16B(&Bt[cur][waveN + ni * 16 + mlane][f * 32 + kbA], &Bt[cur][waveN + ni * 16 + mlane][f * 32 + kbA + 16]);
#pragma unroll
      for (int mi = 0; mi < 2; ++mi)
#pragma unroll
        for (int ni = 0; ni < 4; ++ni)
          acc[mi][ni] = wmma_f16(afrag[mi], bfrag[ni], acc[mi][ni]);
    }

    if (t + 1 < NT) storeTiles(cur ^ 1);
    __syncthreads();
  }

  // Epilogue: C layout row = r + 8*(lane>=16), col = lane%16 within 16x16 tile
  const int rowHalf = (lane >> 4) * 8;
#pragma unroll
  for (int ni = 0; ni < 4; ++ni) {
    const int col = bN + waveN + ni * 16 + mlane;
    const float bval = bias[col];
#pragma unroll
    for (int mi = 0; mi < 2; ++mi) {
#pragma unroll
      for (int r = 0; r < 8; ++r) {
        const size_t row = (size_t)(bM + waveM + mi * 16 + r + rowHalf);
        const float val = acc[mi][ni][r] + bval;
        if constexpr (OUT_F32)
          ((float*)Cv)[row * N + col] = val;
        else
          ((_Float16*)Cv)[row * N + col] = (_Float16)val;
      }
    }
  }
}

// ---------------------------------------------------------------------------
// Flash attention, one (head, 64-query-row) block per workgroup.
// 128 threads = 4 waves, each wave owns 16 query rows. Keys tiled by 32 with
// ping-pong LDS staging of K (row-major) and V (transposed): one barrier per
// key tile, next tile's global loads overlap with score/softmax/PV compute.
// Q is parked in LDS in A-fragment layout (reloaded per tile) to keep the
// persistent register footprint low enough to avoid scratch spills.
// ---------------------------------------------------------------------------
__global__ __launch_bounds__(128, 1)
void attn_wmma(const _Float16* __restrict__ Qb, const _Float16* __restrict__ Kb,
               const _Float16* __restrict__ Vb, _Float16* __restrict__ Ob) {
  __shared__ __align__(16) _Float16 Ks[2][32][136];   // [buf][key][d]
  __shared__ __align__(16) _Float16 Vt[2][128][40];   // [buf][d][key]
  __shared__ __align__(16) _Float16 Ps[4][16][40];    // per-wave P re-layout
  __shared__ __align__(16) _Float16 Qs[4][4][32][16]; // [wave][frag][lane][16h]

  const int tid   = threadIdx.x;
  const int lane  = tid & 31;
  const int wave  = tid >> 5;
  const int h     = blockIdx.y;
  const int qbase = blockIdx.x * 64 + wave * 16;
  const int mlane = lane & 15;
  const int hi    = lane >> 4;
  const int kbA   = hi * 8;
  const int kbB   = hi * 16;

  // Load Q fragments (A layout) once and park them in per-wave LDS:
  // each lane's fragment is a contiguous 32B record -> reload is 2x b128.
  {
    const _Float16* qrow = Qb + (size_t)(qbase + mlane) * E_DIM + h * D_DIM;
#pragma unroll
    for (int f = 0; f < 4; ++f) {
      const _Float16* p = qrow + f * 32 + kbA;
      uint4* dst = reinterpret_cast<uint4*>(&Qs[wave][f][lane][0]);
      dst[0] = *reinterpret_cast<const uint4*>(p);
      dst[1] = *reinterpret_cast<const uint4*>(p + 16);
    }
  }

  const v8f vzero = {0.f, 0.f, 0.f, 0.f, 0.f, 0.f, 0.f, 0.f};
  v8f oacc[8];
#pragma unroll
  for (int j = 0; j < 8; ++j) oacc[j] = vzero;
  float mrow[8], lrow[8];
#pragma unroll
  for (int r = 0; r < 8; ++r) { mrow[r] = -__builtin_inff(); lrow[r] = 0.f; }

  const float scale = 0.08838834764831845f;  // 1/sqrt(128)
  const float L2E   = 1.4426950408889634f;

  const int sRow = tid >> 2;        // 0..31 (key within tile)
  const int sCol = (tid & 3) * 32;  // 0,32,64,96 (d)

  // Prefetched K/V tiles kept packed (16 VGPRs each).
  uint4 kReg[4];
  union VPack { uint4 q[4]; _Float16 h[32]; } vReg;

  auto loadKV = [&](int kt) {
    const uint4* kv =
        reinterpret_cast<const uint4*>(Kb + (size_t)(kt + sRow) * E_DIM + h * D_DIM + sCol);
#pragma unroll
    for (int i = 0; i < 4; ++i) kReg[i] = kv[i];
    const uint4* vv =
        reinterpret_cast<const uint4*>(Vb + (size_t)(kt + sRow) * E_DIM + h * D_DIM + sCol);
#pragma unroll
    for (int i = 0; i < 4; ++i) vReg.q[i] = vv[i];
  };
  auto storeKV = [&](int buf) {
    uint4* kd = reinterpret_cast<uint4*>(&Ks[buf][sRow][sCol]);
#pragma unroll
    for (int i = 0; i < 4; ++i) kd[i] = kReg[i];
#pragma unroll
    for (int i = 0; i < 32; ++i) Vt[buf][sCol + i][sRow] = vReg.h[i];
  };

  loadKV(0);
  storeKV(0);
  __syncthreads();

  const int NKT = S_DIM >> 5;  // S / 32
  for (int t = 0; t < NKT; ++t) {
    const int cur = t & 1;
    const int kt  = t * 32;
    if (t + 1 < NKT) loadKV(kt + 32);  // overlap with compute below
    if (t + 2 < NKT) {
      __builtin_prefetch(Kb + (size_t)(kt + 64 + sRow) * E_DIM + h * D_DIM + sCol, 0, 3);
      __builtin_prefetch(Vb + (size_t)(kt + 64 + sRow) * E_DIM + h * D_DIM + sCol, 0, 3);
    }

    // scores S = Q * K^T : two 16x16 tiles over the 32 keys
    v8f s0acc = vzero, s1acc = vzero;
#pragma unroll
    for (int f = 0; f < 4; ++f) {
      v16h qa = load_frag_32B(&Qs[wave][f][lane][0]);
      v16h b0 = load_frag_2x16B(&Ks[cur][mlane][f * 32 + kbA], &Ks[cur][mlane][f * 32 + kbA + 16]);
      v16h b1 = load_frag_2x16B(&Ks[cur][16 + mlane][f * 32 + kbA], &Ks[cur][16 + mlane][f * 32 + kbA + 16]);
      s0acc = wmma_f16(qa, b0, s0acc);
      s1acc = wmma_f16(qa, b1, s1acc);
    }

    // online softmax: rows live across the 16-lane half-wave.
    // Probabilities go straight to LDS (f16) so no p-arrays stay live.
#pragma unroll
    for (int r = 0; r < 8; ++r) {
      const float s0 = s0acc[r] * scale;
      const float s1 = s1acc[r] * scale;
      float mx = fmaxf(s0, s1);
#pragma unroll
      for (int off = 1; off < 16; off <<= 1)
        mx = fmaxf(mx, __shfl_xor(mx, off, 16));
      const float mnew  = fmaxf(mrow[r], mx);
      const float alpha = exp2f((mrow[r] - mnew) * L2E);
      const float p0 = exp2f((s0 - mnew) * L2E);
      const float p1 = exp2f((s1 - mnew) * L2E);
      Ps[wave][r + hi * 8][mlane]      = (_Float16)p0;
      Ps[wave][r + hi * 8][16 + mlane] = (_Float16)p1;
      float psum = p0 + p1;
#pragma unroll
      for (int off = 1; off < 16; off <<= 1)
        psum += __shfl_xor(psum, off, 16);
      lrow[r] = lrow[r] * alpha + psum;
      mrow[r] = mnew;
#pragma unroll
      for (int j = 0; j < 8; ++j) oacc[j][r] *= alpha;
    }
    asm volatile("s_wait_dscnt 0" ::: "memory");

    v16h pf;
    {
      const _Float16* p = &Ps[wave][mlane][kbA];
      pf = load_frag_2x16B(p, p + 16);
    }
    // O += P * V over all 8 d-tiles
#pragma unroll
    for (int j = 0; j < 8; ++j) {
      v16h b = load_frag_2x16B(&Vt[cur][j * 16 + mlane][kbA], &Vt[cur][j * 16 + mlane][kbA + 16]);
      oacc[j] = wmma_f16(pf, b, oacc[j]);
    }

    if (t + 1 < NKT) storeKV(cur ^ 1);
    __syncthreads();
  }

  // normalize and store (f16, [S, E] with head slice at h*128)
#pragma unroll
  for (int r = 0; r < 8; ++r) {
    const float inv = 1.0f / lrow[r];
    const size_t row = (size_t)(qbase + r + hi * 8);
    _Float16* dst = Ob + row * E_DIM + h * D_DIM + mlane;
#pragma unroll
    for (int j = 0; j < 8; ++j)
      dst[j * 16] = (_Float16)(oacc[j][r] * inv);
  }
}

extern "C" void kernel_launch(void* const* d_in, const int* in_sizes, int n_in,
                              void* d_out, int out_size, void* d_ws, size_t ws_size,
                              hipStream_t stream) {
  (void)in_sizes; (void)n_in; (void)out_size; (void)ws_size;
  const float* x  = (const float*)d_in[0];
  const float* Wq = (const float*)d_in[1];
  const float* bq = (const float*)d_in[2];
  const float* Wk = (const float*)d_in[3];
  const float* bk = (const float*)d_in[4];
  const float* Wv = (const float*)d_in[5];
  const float* bv = (const float*)d_in[6];
  const float* Wo = (const float*)d_in[7];
  const float* bo = (const float*)d_in[8];

  const size_t elems = (size_t)S_DIM * E_DIM;
  _Float16* Qb = (_Float16*)d_ws;
  _Float16* Kb = Qb + elems;
  _Float16* Vb = Kb + elems;
  _Float16* Ob = Vb + elems;

  dim3 gGemm(E_DIM / 128, S_DIM / 128);
  gemm_wmma<float, false><<<gGemm, 256, 0, stream>>>(x, Wq, bq, (void*)Qb, S_DIM, E_DIM, E_DIM);
  gemm_wmma<float, false><<<gGemm, 256, 0, stream>>>(x, Wk, bk, (void*)Kb, S_DIM, E_DIM, E_DIM);
  gemm_wmma<float, false><<<gGemm, 256, 0, stream>>>(x, Wv, bv, (void*)Vb, S_DIM, E_DIM, E_DIM);

  dim3 gAttn(S_DIM / 64, H_DIM);
  attn_wmma<<<gAttn, 128, 0, stream>>>(Qb, Kb, Vb, Ob);

  gemm_wmma<_Float16, true><<<gGemm, 256, 0, stream>>>(Ob, Wo, bo, d_out, S_DIM, E_DIM, E_DIM);
}
